// MIGCProcessor_76751065580159
// MI455X (gfx1250) — compile-verified, matrix operand order checked by reading, not run on hardware
//
#include <hip/hip_runtime.h>
#include <hip/hip_bf16.h>
#include <math.h>

typedef _Float16 v16h __attribute__((ext_vector_type(16)));
typedef float    v8f  __attribute__((ext_vector_type(8)));

#define HEADS 8
#define DHEAD 40
#define DPAD  64     // head dim padded to 64 (2 x K32 steps), zeros beyond 40
#define HSTR  (HEADS * DPAD)   // 512 halves per padded row
#define HW    4096
#define C     320
#define CX    768
#define L     77
#define NB    10     // batches after the repeat (1 uncond + 9 cond)
#define NINST 8
#define MKV   896    // 770 K/V rows padded up to GEMM M-tile (128) multiple
#define KPAD  80     // keys padded to 80 (5 x 16) for WMMA
#define PSTR  96     // P-tile LDS row stride in halves (3 x K32 steps)

// ---------------- f32 -> f16 convert (row-major copy) ----------------
__global__ void k_cvt_f16(const float* __restrict__ src, _Float16* __restrict__ dst, int n) {
  int i = blockIdx.x * blockDim.x + threadIdx.x;
  if (i < n) dst[i] = (_Float16)src[i];
}

// ---------------- W (K x N) -> Wt (N x K) in f16 ----------------
__global__ void k_transpose_f16(const float* __restrict__ W, _Float16* __restrict__ Wt, int K, int N) {
  int i = blockIdx.x * blockDim.x + threadIdx.x;
  if (i >= K * N) return;
  int k = i / N, n = i % N;
  Wt[(size_t)n * K + k] = (_Float16)W[(size_t)k * N + n];
}

// ---------------- Q (m, h*40+d) -> Qp (m, h*64+d) zero-padded ----------------
__global__ void k_qpad(const _Float16* __restrict__ Qh, _Float16* __restrict__ Qp) {
  int idx = blockIdx.x * blockDim.x + threadIdx.x;     // 2*HW*HSTR
  if (idx >= 2 * HW * HSTR) return;
  int hd = idx & (HSTR - 1);
  int m  = idx >> 9;
  int h = hd >> 6, d = hd & 63;
  Qp[idx] = (d < DHEAD) ? Qh[(size_t)m * C + h * DHEAD + d] : (_Float16)0.f;
}

// ---------------- K (b*77+l, h*40+d) -> Kp (b*80+key, h*64+d) zero-padded ----------------
__global__ void k_kpad(const _Float16* __restrict__ Kh, _Float16* __restrict__ Kp) {
  int idx = blockIdx.x * blockDim.x + threadIdx.x;     // NB*KPAD*HSTR
  if (idx >= NB * KPAD * HSTR) return;
  int hd = idx & (HSTR - 1);
  int bk = idx >> 9;
  int b = bk / KPAD, key = bk % KPAD;
  int h = hd >> 6, d = hd & 63;
  Kp[idx] = (key < L && d < DHEAD)
                ? Kh[(size_t)(b * L + key) * C + h * DHEAD + d]
                : (_Float16)0.f;
}

// ---------------- V (b,l,c) f16 -> Vt (b,c,keypad80) f16, zero key>=77 ----------------
__global__ void k_vt(const _Float16* __restrict__ Vtmp, _Float16* __restrict__ Vt) {
  int idx = blockIdx.x * blockDim.x + threadIdx.x;    // NB*C*KPAD
  if (idx >= NB * C * KPAD) return;
  int l = idx % KPAD;
  int bc = idx / KPAD;
  int b = bc / C, c = bc % C;
  Vt[idx] = (l < L) ? Vtmp[(size_t)(b * L + l) * C + c] : (_Float16)0.f;
}

// ---------------- WMMA GEMM: C[MxN] = A[MxK](f16) * Bt[NxK](f16, =B^T) ----------------
// block = 256 threads = 8 waves; block tile 128(M) x 64(N); each wave: 16 x 64.
template <typename OT>
__global__ void k_gemm_wmma(const _Float16* __restrict__ A, const _Float16* __restrict__ Bt,
                            OT* __restrict__ Cm, int M, int N, int K) {
  const int lane = threadIdx.x & 31;
  const int wave = threadIdx.x >> 5;
  const int m0 = blockIdx.x * 128 + wave * 16;
  const int n0 = blockIdx.y * 64;
  const int ml = lane & 15;
  const int kb = (lane >> 4) << 3;           // lanes 16..31 hold K+8 per ISA A-layout

  v8f acc[4] = {};
  const _Float16* Arow = A + (size_t)(m0 + ml) * K;

  for (int kk = 0; kk < K; kk += 32) {
    union { v16h v; unsigned u[8]; } fa;
#pragma unroll
    for (int p = 0; p < 8; ++p) {
      int k = kk + ((p & 4) << 2) + kb + ((p & 3) << 1);
      fa.u[p] = *(const unsigned*)(Arow + k);
    }
#pragma unroll
    for (int t = 0; t < 4; ++t) {
      const _Float16* Brow = Bt + (size_t)(n0 + t * 16 + ml) * K;
      union { v16h v; unsigned u[8]; } fb;
#pragma unroll
      for (int p = 0; p < 8; ++p) {
        int k = kk + ((p & 4) << 2) + kb + ((p & 3) << 1);
        fb.u[p] = *(const unsigned*)(Brow + k);
      }
      acc[t] = __builtin_amdgcn_wmma_f32_16x16x32_f16(false, fa.v, false, fb.v,
                                                      (short)0, acc[t], false, false);
    }
  }

#pragma unroll
  for (int t = 0; t < 4; ++t) {
#pragma unroll
    for (int r = 0; r < 8; ++r) {
      int m = m0 + r + ((lane >> 4) << 3);
      int n = n0 + t * 16 + (lane & 15);
      Cm[(size_t)m * N + n] = (OT)acc[t][r];
    }
  }
}

// ---------------- WMMA flash-attention: one wave per (batch, head, 16-query tile) ----------------
// S(16x80) = Qp(16x64) @ Kp^T   -> softmax (16-lane shfl reductions) -> P staged via LDS
// O(16x48pad) = P(16x96pad) @ V -> store d < 40
__global__ void k_attn_wmma(const _Float16* __restrict__ Qp, const _Float16* __restrict__ Kp,
                            const _Float16* __restrict__ Vt, float* __restrict__ O) {
  __shared__ __align__(16) _Float16 plds[8][16][PSTR];

  const int lane = threadIdx.x & 31;
  const int wave = threadIdx.x >> 5;
  const int w = blockIdx.x * 8 + wave;       // global wave id: 80 bh * 256 qtiles
  const int qt = w & 255;
  const int bh = w >> 8;
  const int b = bh >> 3, h = bh & 7;
  const int bq = b ? 1 : 0;
  const int m0 = qt * 16;
  const int ml = lane & 15;
  const int nl = lane & 15;
  const int hl = lane >> 4;
  const int kb = hl << 3;
  const float scale = 0.15811388300841898f;  // 1/sqrt(40)

  // ---- scores: 5 key tiles x 2 K-steps (D padded to 64, zeros in memory) ----
  v8f sAcc[5] = {};
  const _Float16* Qrow = Qp + ((size_t)(bq * HW + m0 + ml)) * HSTR + h * DPAD;
#pragma unroll
  for (int ks = 0; ks < 2; ++ks) {
    const int kk = ks * 32;
    union { v16h v; unsigned u[8]; } fa;
#pragma unroll
    for (int p = 0; p < 8; ++p) {
      int k = kk + ((p & 4) << 2) + kb + ((p & 3) << 1);
      fa.u[p] = *(const unsigned*)(Qrow + k);
    }
#pragma unroll
    for (int t = 0; t < 5; ++t) {
      const _Float16* Krow = Kp + ((size_t)(b * KPAD) + t * 16 + ml) * HSTR + h * DPAD;
      union { v16h v; unsigned u[8]; } fb;
#pragma unroll
      for (int p = 0; p < 8; ++p) {
        int k = kk + ((p & 4) << 2) + kb + ((p & 3) << 1);
        fb.u[p] = *(const unsigned*)(Krow + k);
      }
      sAcc[t] = __builtin_amdgcn_wmma_f32_16x16x32_f16(false, fa.v, false, fb.v,
                                                       (short)0, sAcc[t], false, false);
    }
  }

  // ---- row softmax; write normalized P (f16) to LDS, zero pad keys 80..95 ----
  _Float16* Pl = &plds[wave][0][0];
#pragma unroll
  for (int r = 0; r < 8; ++r) {
    float mx = -1e30f;
#pragma unroll
    for (int t = 0; t < 5; ++t) {
      int n = t * 16 + nl;
      if (n < L) mx = fmaxf(mx, sAcc[t][r] * scale);
    }
    mx = fmaxf(mx, __shfl_xor(mx, 1, 32));
    mx = fmaxf(mx, __shfl_xor(mx, 2, 32));
    mx = fmaxf(mx, __shfl_xor(mx, 4, 32));
    mx = fmaxf(mx, __shfl_xor(mx, 8, 32));
    float p[5], sum = 0.f;
#pragma unroll
    for (int t = 0; t < 5; ++t) {
      int n = t * 16 + nl;
      float e = (n < L) ? __expf(sAcc[t][r] * scale - mx) : 0.f;
      p[t] = e; sum += e;
    }
    sum += __shfl_xor(sum, 1, 32);
    sum += __shfl_xor(sum, 2, 32);
    sum += __shfl_xor(sum, 4, 32);
    sum += __shfl_xor(sum, 8, 32);
    float inv = 1.f / sum;
    int m = r + (hl << 3);
#pragma unroll
    for (int t = 0; t < 5; ++t) Pl[m * PSTR + t * 16 + nl] = (_Float16)(p[t] * inv);
    Pl[m * PSTR + 80 + nl] = (_Float16)0.f;                        // zero keys 80..95
  }
  __builtin_amdgcn_wave_barrier();

  // ---- O = P @ V : 3 N-tiles (40 -> 48) x 3 K-steps (80 -> 96) ----
  v8f oAcc[3] = {};
  const _Float16* ArowL = Pl + ml * PSTR;
#pragma unroll
  for (int ks = 0; ks < 3; ++ks) {
    const int kk = ks * 32;
    union { v16h v; unsigned u[8]; } fa;
#pragma unroll
    for (int p = 0; p < 8; ++p) {
      int k = kk + ((p & 4) << 2) + kb + ((p & 3) << 1);
      fa.u[p] = *(const unsigned*)(ArowL + k);                     // LDS read (P zero k>=77)
    }
#pragma unroll
    for (int t = 0; t < 3; ++t) {
      const _Float16* Brow = Vt + ((size_t)(b * C) + h * DHEAD + t * 16 + ml) * KPAD;
      union { v16h v; unsigned u[8]; } fb;
#pragma unroll
      for (int p = 0; p < 8; ++p) {
        int k = kk + ((p & 4) << 2) + kb + ((p & 3) << 1);
        fb.u[p] = *(const unsigned*)(Brow + k);
      }
      oAcc[t] = __builtin_amdgcn_wmma_f32_16x16x32_f16(false, fa.v, false, fb.v,
                                                       (short)0, oAcc[t], false, false);
    }
  }

#pragma unroll
  for (int t = 0; t < 3; ++t) {
#pragma unroll
    for (int r = 0; r < 8; ++r) {
      int n = t * 16 + nl;
      if (n < DHEAD) {
        int m = m0 + r + (hl << 3);
        O[((size_t)(b * HW) + m) * C + h * DHEAD + n] = oAcc[t][r];
      }
    }
  }
}

// ---------------- bilinear-antialias 8x downsample of box masks (separable) ----------------
__device__ __forceinline__ float tri_factor(int o, float lo, float hi) {
  float center = 8.0f * (float)o + 3.5f;
  float ks = 0.f, ms = 0.f;
  int s0 = 8 * o - 4;
#pragma unroll
  for (int t = 0; t < 16; ++t) {
    int s = s0 + t;
    if (s < 0 || s >= 512) continue;
    float kv = 1.0f - fabsf((float)s - center) * 0.125f;
    if (kv <= 0.f) continue;
    ks += kv;
    float fs = (float)s;
    if (fs >= lo && fs < hi) ms += kv;
  }
  return (ks > 0.f) ? ms / ks : 0.f;
}

__global__ void k_mask(const float* __restrict__ bboxes, float* __restrict__ W9,
                       float* __restrict__ Wsum) {
  int hw = blockIdx.x * blockDim.x + threadIdx.x;
  if (hw >= HW) return;
  int r = hw >> 6, c = hw & 63;
  float sum = 0.1f;
  W9[hw] = 0.1f;
  for (int i = 0; i < NINST; ++i) {
    float x0 = bboxes[i * 4 + 0], y0 = bboxes[i * 4 + 1];
    float x1 = bboxes[i * 4 + 2], y1 = bboxes[i * 4 + 3];
    float wlo = floorf(512.f * x0), whi = floorf(512.f * x1);
    float hlo = floorf(512.f * y0), hhi = floorf(512.f * y1);
    float w = tri_factor(r, hlo, hhi) * tri_factor(c, wlo, whi) * 10.f;
    W9[(size_t)(i + 1) * HW + hw] = w;
    sum += w;
  }
  Wsum[hw] = sum;
}

// ---------------- fold weighted fusion BEFORE the Wo GEMM (linearity) ----------------
__global__ void k_reduce(const float* __restrict__ O, const float* __restrict__ W9,
                         _Float16* __restrict__ X) {
  int idx = blockIdx.x * blockDim.x + threadIdx.x;      // HW*C
  if (idx >= HW * C) return;
  int hw = idx / C;
  X[idx] = (_Float16)O[idx];
  float a = 0.f;
#pragma unroll
  for (int i = 0; i < 9; ++i)
    a += W9[(size_t)i * HW + hw] * O[(size_t)(i + 1) * HW * C + idx];
  X[(size_t)HW * C + idx] = (_Float16)a;
}

// ---------------- epilogue: bias + normalization ----------------
__global__ void k_epilogue(const float* __restrict__ Y, const float* __restrict__ Wsum,
                           const float* __restrict__ bo, float* __restrict__ out) {
  int idx = blockIdx.x * blockDim.x + threadIdx.x;      // 2*HW*C
  if (idx >= 2 * HW * C) return;
  int c = idx % C;
  if (idx < HW * C) {
    out[idx] = Y[idx] + bo[c];
  } else {
    int hw = (idx - HW * C) / C;
    float ws = Wsum[hw];
    out[idx] = (Y[idx] + ws * bo[c]) / (ws + 1e-6f);
  }
}

extern "C" void kernel_launch(void* const* d_in, const int* in_sizes, int n_in,
                              void* d_out, int out_size, void* d_ws, size_t ws_size,
                              hipStream_t stream) {
  (void)in_sizes; (void)n_in; (void)out_size; (void)ws_size;
  const float* hs  = (const float*)d_in[0];   // (2, 4096, 320)
  const float* ehs = (const float*)d_in[1];   // (11, 77, 768) — use first 10
  const float* bb  = (const float*)d_in[2];   // (1, 8, 4)
  const float* Wq  = (const float*)d_in[3];
  const float* Wk  = (const float*)d_in[4];
  const float* Wv  = (const float*)d_in[5];
  const float* Wo  = (const float*)d_in[6];
  const float* bo  = (const float*)d_in[7];
  float* out = (float*)d_out;                 // (2, 4096, 320)

  char* wsb = (char*)d_ws;
  size_t off = 0;
  auto alloc = [&](size_t bytes) -> void* {
    void* p = wsb + off;
    off += (bytes + 255) & ~(size_t)255;
    return p;
  };
  _Float16* hs_h  = (_Float16*)alloc((size_t)2 * HW * C * 2);
  _Float16* ehs_h = (_Float16*)alloc((size_t)MKV * CX * 2);          // 770 rows valid
  _Float16* Wq_t  = (_Float16*)alloc((size_t)C * C * 2);
  _Float16* Wo_t  = (_Float16*)alloc((size_t)C * C * 2);
  _Float16* Wk_t  = (_Float16*)alloc((size_t)C * CX * 2);
  _Float16* Wv_t  = (_Float16*)alloc((size_t)C * CX * 2);
  _Float16* Qh    = (_Float16*)alloc((size_t)2 * HW * C * 2);
  _Float16* Qp    = (_Float16*)alloc((size_t)2 * HW * HSTR * 2);
  _Float16* Kh    = (_Float16*)alloc((size_t)MKV * C * 2);
  _Float16* Kp    = (_Float16*)alloc((size_t)NB * KPAD * HSTR * 2);
  _Float16* Vtmp  = (_Float16*)alloc((size_t)MKV * C * 2);
  _Float16* Vt    = (_Float16*)alloc(((size_t)NB * C + 16) * KPAD * 2);
  float*    O     = (float*)   alloc((size_t)NB * HW * C * 4);
  float*    W9    = (float*)   alloc((size_t)9 * HW * 4);
  float*    Wsum  = (float*)   alloc((size_t)HW * 4);
  _Float16* X     = (_Float16*)alloc((size_t)2 * HW * C * 2);
  float*    Y     = (float*)   alloc((size_t)2 * HW * C * 4);

  const int TB = 256;
  // convert activations to f16
  k_cvt_f16<<<dim3((2 * HW * C + TB - 1) / TB), dim3(TB), 0, stream>>>(hs, hs_h, 2 * HW * C);
  k_cvt_f16<<<dim3((NB * L * CX + TB - 1) / TB), dim3(TB), 0, stream>>>(ehs, ehs_h, NB * L * CX);
  // transpose weights to f16 B^T operands
  k_transpose_f16<<<dim3((C * C + TB - 1) / TB), dim3(TB), 0, stream>>>(Wq, Wq_t, C, C);
  k_transpose_f16<<<dim3((C * C + TB - 1) / TB), dim3(TB), 0, stream>>>(Wo, Wo_t, C, C);
  k_transpose_f16<<<dim3((CX * C + TB - 1) / TB), dim3(TB), 0, stream>>>(Wk, Wk_t, CX, C);
  k_transpose_f16<<<dim3((CX * C + TB - 1) / TB), dim3(TB), 0, stream>>>(Wv, Wv_t, CX, C);
  // projections (all WMMA)
  k_gemm_wmma<_Float16><<<dim3(2 * HW / 128, C / 64), dim3(TB), 0, stream>>>(
      hs_h, Wq_t, Qh, 2 * HW, C, C);
  k_gemm_wmma<_Float16><<<dim3(MKV / 128, C / 64), dim3(TB), 0, stream>>>(
      ehs_h, Wk_t, Kh, MKV, C, CX);
  k_gemm_wmma<_Float16><<<dim3(MKV / 128, C / 64), dim3(TB), 0, stream>>>(
      ehs_h, Wv_t, Vtmp, MKV, C, CX);
  // repack into zero-padded attention layouts (branch-free coalesced fragments)
  k_qpad<<<dim3(2 * HW * HSTR / TB), dim3(TB), 0, stream>>>(Qh, Qp);
  k_kpad<<<dim3(NB * KPAD * HSTR / TB), dim3(TB), 0, stream>>>(Kh, Kp);
  k_vt<<<dim3((NB * C * KPAD + TB - 1) / TB), dim3(TB), 0, stream>>>(Vtmp, Vt);
  // WMMA flash attention: 80 bh * 256 qtiles waves, 8 waves/block
  k_attn_wmma<<<dim3(NB * HEADS * (HW / 16) / 8), dim3(TB), 0, stream>>>(Qp, Kp, Vt, O);
  // masks + linear fusion fold + output projection (WMMA) + epilogue
  k_mask<<<dim3(HW / TB), dim3(TB), 0, stream>>>(bb, W9, Wsum);
  k_reduce<<<dim3(HW * C / TB), dim3(TB), 0, stream>>>(O, W9, X);
  k_gemm_wmma<float><<<dim3(2 * HW / 128, C / 64), dim3(TB), 0, stream>>>(
      X, Wo_t, Y, 2 * HW, C, C);
  k_epilogue<<<dim3(2 * HW * C / TB), dim3(TB), 0, stream>>>(Y, Wsum, bo, out);
}
